// RCNN_31190052503701
// MI455X (gfx1250) — matrix operand order repaired and checked by reading, not compiled, not run on hardware
//
#include <hip/hip_runtime.h>
#include <hip/hip_bf16.h>

typedef __attribute__((ext_vector_type(16))) _Float16 v16h;
typedef __attribute__((ext_vector_type(8)))  _Float16 v8h;
typedef __attribute__((ext_vector_type(8)))  float    v8f;
typedef __attribute__((ext_vector_type(4)))  float    v4f;

// ---------------------------------------------------------------------------
// Weight repack: [Cout][Cin][3][3] f32 -> [Cout][Kpad] f16, K = j*Cin + c
// (tap-major, channel-innermost), each 32-K block kept in contiguous k order
// (lanes 0-15: k 0-15, lanes 16-31: k 16-31) so an A fragment is ONE v16h load.
// ---------------------------------------------------------------------------
__global__ void pack_wgt_f16(const float* __restrict__ w, _Float16* __restrict__ dst,
                             int Cout, int CIN, int Kpad) {
    size_t i = (size_t)blockIdx.x * blockDim.x + threadIdx.x;
    if (i >= (size_t)Cout * (size_t)Kpad) return;
    int kk = (int)(i % (size_t)Kpad);
    int co = (int)(i / (size_t)Kpad);
    int t = kk & 31;
    int tp = t;  // identity (contiguous K order)
    int kor = (kk & ~31) | tp;
    int K = 9 * CIN;
    _Float16 v = (_Float16)0.0f;
    if (kor < K) {
        int j = kor / CIN, c = kor - j * CIN;
        v = (_Float16)w[((size_t)co * CIN + c) * 9 + j];
    }
    dst[i] = v;
}

// Zero the 1-pixel border of a padded NHWC [4][Hp][Hp][C] f16 buffer.
__global__ void zero_border_nhwc(_Float16* __restrict__ buf, int C, int Hp) {
    size_t per = 4 * (size_t)Hp * C;             // 2 rows + 2 cols (corners twice, harmless)
    size_t i = (size_t)blockIdx.x * blockDim.x + threadIdx.x;
    if (i >= 4 * per) return;
    int n = (int)(i / per);
    int t = (int)(i % per);
    int c = t % C;
    int q = t / C;                               // [0, 4*Hp)
    int row, col;
    if (q < 2 * Hp) { row = (q < Hp) ? 0 : (Hp - 1); col = q % Hp; }
    else            { q -= 2 * Hp; col = (q < Hp) ? 0 : (Hp - 1); row = q % Hp; }
    buf[(((size_t)n * Hp + row) * Hp + col) * C + c] = (_Float16)0.0f;
}

// Input f32 NCHW [4][3][224][224] -> f16 padded NHWC [4][226][226][3] interior.
__global__ void cvt_pad_in(const float* __restrict__ src, _Float16* __restrict__ dst) {
    size_t i = (size_t)blockIdx.x * blockDim.x + threadIdx.x;
    if (i >= (size_t)12 * 224 * 224) return;
    int w = (int)(i % 224);
    int h = (int)((i / 224) % 224);
    int c = (int)((i / (224 * 224)) % 3);
    int n = (int)(i / (3 * 224 * 224));
    dst[(((size_t)n * 226 + h + 1) * 226 + w + 1) * 3 + c] = (_Float16)src[i];
}

// 2x2 stride-2 maxpool, padded NHWC ([4][H+2][H+2][C] -> [4][H/2+2][H/2+2][C])
__global__ void maxpool2_nhwc(const _Float16* __restrict__ in, _Float16* __restrict__ out,
                              int C, int H) {
    const int Ho = H >> 1, Hp = H + 2, Hop = Ho + 2;
    size_t i = (size_t)blockIdx.x * blockDim.x + threadIdx.x;
    if (i >= (size_t)4 * Ho * Ho * C) return;
    int c  = (int)(i % C);
    int wo = (int)((i / C) % Ho);
    int ho = (int)((i / ((size_t)C * Ho)) % Ho);
    int n  = (int)(i / ((size_t)C * Ho * Ho));
    const _Float16* p = in + (((size_t)n * Hp + 2 * ho + 1) * Hp + 2 * wo + 1) * C + c;
    const size_t rs = (size_t)Hp * C;
    float m = fmaxf(fmaxf((float)p[0], (float)p[C]), fmaxf((float)p[rs], (float)p[rs + C]));
    out[(((size_t)n * Hop + ho + 1) * Hop + wo + 1) * C + c] = (_Float16)m;
}

// ---------------------------------------------------------------------------
// Fast 3x3 conv, Cin % 32 == 0, padded NHWC: implicit GEMM, one wave computes
// a 32(Cout) x 16(pixel) tile. Per K-step: 2 v16h A loads + 1 v16h B load +
// 2 x v_wmma_f32_16x16x32_f16. No index math, no bounds checks in the loop.
// ---------------------------------------------------------------------------
template <int CIN, int H>
__global__ __launch_bounds__(256) void conv3x3_wmma_t(
    const _Float16* __restrict__ act,   // padded NHWC [4][H+2][H+2][CIN]
    const _Float16* __restrict__ wgt,   // [Cout][9*CIN], lane-order packed
    const float* __restrict__ bias,
    _Float16* __restrict__ out,         // padded NHWC [4][H+2][H+2][Cout]
    int Cout)
{
    constexpr int Hp = H + 2;
    constexpr int K  = 9 * CIN;
    const int lane = threadIdx.x & 31;
    const int wave = threadIdx.x >> 5;
    const int total = 4 * H * H;
    const int pixBase = (blockIdx.x * 8 + wave) * 16;
    if (pixBase >= total) return;
    const int coutBase = blockIdx.y * 32;

    const int l15 = lane & 15;
    int p = pixBase + l15;
    if (p >= total) p = total - 1;
    const int n   = p / (H * H);
    const int rem = p - n * (H * H);
    const int oh  = rem / H;
    const int ow  = rem - oh * H;

    const int sel = (lane < 16) ? 0 : 16;   // fragment half selector (A packed, B K-split)
    const _Float16* w0 = wgt + (size_t)(coutBase + l15) * K + sel;
    const _Float16* w1 = w0 + (size_t)16 * K;
    const _Float16* bp = act + (((size_t)n * Hp + oh) * Hp + ow) * CIN + sel;

    v8f acc0 = {}, acc1 = {};
#pragma unroll
    for (int j = 0; j < 9; ++j) {
        const int r = j / 3, s = j - 3 * (j / 3);          // compile-time after unroll
        const _Float16* bj = bp + (size_t)(r * Hp + s) * CIN;
        for (int c0 = 0; c0 < CIN; c0 += 32) {
            const int k0 = j * CIN + c0;
            v16h a0 = *(const v16h*)(w0 + k0);             // one 32B load (lane-packed)
            v16h a1 = *(const v16h*)(w1 + k0);
            v16h b  = *(const v16h*)(bj + c0);             // one 32B load (NHWC)
            acc0 = __builtin_amdgcn_wmma_f32_16x16x32_f16(false, a0, false, b, (short)0, acc0,
                                                          false, false);
            acc1 = __builtin_amdgcn_wmma_f32_16x16x32_f16(false, a1, false, b, (short)0, acc1,
                                                          false, false);
        }
    }

    if (pixBase + l15 < total) {
        const int mOff = (lane < 16) ? 0 : 8;
        _Float16* op = out + (((size_t)n * Hp + oh + 1) * Hp + (ow + 1)) * Cout
                       + coutBase + mOff;
        v8h o0, o1;
#pragma unroll
        for (int j = 0; j < 8; ++j) {
            o0[j] = (_Float16)fmaxf(acc0[j] + bias[coutBase + mOff + j], 0.0f);
            o1[j] = (_Float16)fmaxf(acc1[j] + bias[coutBase + mOff + 16 + j], 0.0f);
        }
        *(v8h*)(op)      = o0;                             // contiguous Cout, 16B aligned
        *(v8h*)(op + 16) = o1;
    }
}

// ---------------------------------------------------------------------------
// Layer 0: Cin=3, K=27 (padded to 32), NHWC -> one WMMA per 16x16 tile.
// ---------------------------------------------------------------------------
__device__ __forceinline__ _Float16 conv0_elem(const _Float16* bp, int k) {
    if (k >= 27) return (_Float16)0.0f;
    const int j = k / 3, c = k - 3 * j;     // k = j*3 + c
    const int r = j / 3, s = j - 3 * r;
    return bp[(r * 226 + s) * 3 + c];       // immediate after constant folding
}

__global__ __launch_bounds__(256) void conv0_wmma(
    const _Float16* __restrict__ act,   // padded NHWC [4][226][226][3]
    const _Float16* __restrict__ wgt,   // [64][32] lane-order packed
    const float* __restrict__ bias,
    _Float16* __restrict__ out)         // padded NHWC [4][226][226][64]
{
    constexpr int H = 224, Hp = 226, Cout = 64;
    const int lane = threadIdx.x & 31;
    const int wave = threadIdx.x >> 5;
    const int total = 4 * H * H;
    const int pixBase = (blockIdx.x * 8 + wave) * 16;
    if (pixBase >= total) return;
    const int coutBase = blockIdx.y * 16;

    const int l15 = lane & 15;
    int p = pixBase + l15;
    if (p >= total) p = total - 1;
    const int n   = p / (H * H);
    const int rem = p - n * (H * H);
    const int oh  = rem / H;
    const int ow  = rem - oh * H;

    const int sel = (lane < 16) ? 0 : 16;
    v16h a = *(const v16h*)(wgt + (size_t)(coutBase + l15) * 32 + sel);
    const _Float16* bp = act + (((size_t)n * Hp + oh) * Hp + ow) * 3;
    v16h b;
    if (lane < 16) {
#pragma unroll
        for (int e = 0; e < 16; ++e) b[e] = conv0_elem(bp, e);
    } else {
#pragma unroll
        for (int e = 0; e < 16; ++e) b[e] = conv0_elem(bp, 16 + e);
    }
    v8f acc = {};
    acc = __builtin_amdgcn_wmma_f32_16x16x32_f16(false, a, false, b, (short)0, acc, false, false);

    if (pixBase + l15 < total) {
        const int mOff = (lane < 16) ? 0 : 8;
        _Float16* op = out + (((size_t)n * Hp + oh + 1) * Hp + (ow + 1)) * Cout
                       + coutBase + mOff;
        v8h o;
#pragma unroll
        for (int j = 0; j < 8; ++j)
            o[j] = (_Float16)fmaxf(acc[j] + bias[coutBase + mOff + j], 0.0f);
        *(v8h*)(op) = o;
    }
}

// ---------------------------------------------------------------------------
// FC GEMM (HBM-bound): out[r][m] = relu(sum_k W[m][k]*X[r][k] + bias[m]).
// f32 weights streamed + converted in-register; one wave per 16x16 tile.
// ---------------------------------------------------------------------------
__global__ __launch_bounds__(256) void fc_wmma(
    const float* __restrict__ Wm, const _Float16* __restrict__ X,
    const float* __restrict__ bias, _Float16* __restrict__ out,
    int Nout, int K, int R, int doRelu)
{
    const int lane = threadIdx.x & 31;
    const int wave = threadIdx.x >> 5;
    const int ntiles = Nout >> 4;
    const int g = blockIdx.x * 8 + wave;
    if (g >= ntiles * (R >> 4)) return;
    const int nt = g % ntiles;
    const int rt = g / ntiles;

    const int l15 = lane & 15;
    const int kbA = (lane < 16) ? 0 : 16;
    const int kbB = (lane < 16) ? 0 : 16;
    const float*    wr = Wm + (size_t)(nt * 16 + l15) * K;
    const _Float16* xr = X  + (size_t)(rt * 16 + l15) * K;

    v8f acc = {};
    for (int k0 = 0; k0 < K; k0 += 32) {
        __builtin_prefetch(wr + k0 + 512, 0, 0);
        v4f f0 = *(const v4f*)(wr + k0 + kbA);
        v4f f1 = *(const v4f*)(wr + k0 + kbA + 4);
        v4f f2 = *(const v4f*)(wr + k0 + kbA + 8);
        v4f f3 = *(const v4f*)(wr + k0 + kbA + 12);
        v16h a;
#pragma unroll
        for (int e = 0; e < 4; ++e) {
            a[e]      = (_Float16)f0[e];
            a[e + 4]  = (_Float16)f1[e];
            a[e + 8]  = (_Float16)f2[e];
            a[e + 12] = (_Float16)f3[e];
        }
        v16h b = *(const v16h*)(xr + k0 + kbB);
        acc = __builtin_amdgcn_wmma_f32_16x16x32_f16(false, a, false, b, (short)0, acc,
                                                     false, false);
    }
    const int mOff = (lane < 16) ? 0 : 8;
    const int col  = rt * 16 + l15;
#pragma unroll
    for (int j = 0; j < 8; ++j) {
        const int m = nt * 16 + j + mOff;
        float v = acc[j] + bias[m];
        if (doRelu) v = fmaxf(v, 0.0f);
        out[(size_t)col * Nout + m] = (_Float16)v;
    }
}

// ---------------------------------------------------------------------------
// Adaptive ROI max-pool 7x7 on padded NHWC feat [4][16][16][512]
// -> pooled [128][25088] f16 (flat index c*49 + oy*7 + ox per ROI).
// Channels innermost in the thread index for coalesced feature reads.
// ---------------------------------------------------------------------------
__global__ void roi_pool_k(const _Float16* __restrict__ feat, const float* __restrict__ rois,
                           const int* __restrict__ ridx, _Float16* __restrict__ pooled)
{
    const int i   = blockIdx.x * blockDim.x + threadIdx.x;
    const int c   = i % 512;
    const int pos = (i / 512) % 49;
    const int r   = i / (512 * 49);
    if (r >= 128) return;
    const int ox = pos % 7, oy = pos / 7;
    const int x1 = (int)floorf(rois[4 * r + 0] * 14.0f);
    const int y1 = (int)floorf(rois[4 * r + 1] * 14.0f);
    const int x2 = (int)ceilf (rois[4 * r + 2] * 14.0f);
    const int y2 = (int)ceilf (rois[4 * r + 3] * 14.0f);
    const int Hr = y2 - y1, Wr = x2 - x1;
    const int hs = y1 + (oy * Hr) / 7;
    const int he = y1 + ((oy + 1) * Hr + 6) / 7;
    const int wlo = x1 + (ox * Wr) / 7;
    const int whi = x1 + ((ox + 1) * Wr + 6) / 7;
    const _Float16* img = feat + (size_t)ridx[r] * 16 * 16 * 512;
    float m = -1e30f;
    for (int h = hs; h < he; ++h)
        for (int w = wlo; w < whi; ++w) {
            float v = (float)img[(((h + 1) * 16) + (w + 1)) * 512 + c];
            m = fmaxf(m, v);
        }
    pooled[(size_t)r * 25088 + c * 49 + oy * 7 + ox] = (_Float16)m;
}

// ---------------------------------------------------------------------------
// Heads: wave32-per-output dots with shuffle reduction.
// d_out = cls_score [128,2] then bbox [128,2,4].
// ---------------------------------------------------------------------------
__global__ __launch_bounds__(256) void heads_k(
    const _Float16* __restrict__ h, const float* __restrict__ cls_w, const float* __restrict__ cls_b,
    const float* __restrict__ bbox_w, const float* __restrict__ bbox_b, float* __restrict__ out)
{
    const int lane = threadIdx.x & 31;
    const int g = blockIdx.x * 8 + (threadIdx.x >> 5);
    const int r = g / 6, j = g % 6;
    if (r >= 128) return;
    const float* wrow = (j < 2) ? (cls_w + (size_t)j * 4096) : (bbox_w + (size_t)(j - 2) * 4096);
    const _Float16* hr = h + (size_t)r * 4096;
    float s = 0.0f;
    for (int k = lane; k < 4096; k += 32) s += (float)hr[k] * wrow[k];
#pragma unroll
    for (int off = 16; off; off >>= 1) s += __shfl_xor(s, off, 32);
    if (lane == 0) {
        if (j < 2) {
            out[r * 2 + j] = s + cls_b[j];
        } else {
            const float v = s + bbox_b[j - 2];
            out[256 + r * 8 + (j - 2)]     = v;
            out[256 + r * 8 + 4 + (j - 2)] = v;
        }
    }
}

// ---------------------------------------------------------------------------
// Host orchestration. d_in order (flattened setup_inputs dict):
//   0 inp, 1 rois, 2 roi_idx, 3..15 conv_ws, 16..28 conv_bs,
//   29 fc1_w, 30 fc1_b, 31 fc2_w, 32 fc2_b, 33 cls_w, 34 cls_b, 35 bbox_w, 36 bbox_b
// ---------------------------------------------------------------------------
extern "C" void kernel_launch(void* const* d_in, const int* in_sizes, int n_in,
                              void* d_out, int out_size, void* d_ws, size_t ws_size,
                              hipStream_t stream)
{
    (void)in_sizes; (void)n_in; (void)out_size; (void)ws_size;

    const float* inp  = (const float*)d_in[0];
    const float* rois = (const float*)d_in[1];
    const int*   ridx = (const int*)d_in[2];
    const float* conv_w[13];
    const float* conv_b[13];
    for (int i = 0; i < 13; ++i) {
        conv_w[i] = (const float*)d_in[3 + i];
        conv_b[i] = (const float*)d_in[16 + i];
    }
    const float* fc1_w  = (const float*)d_in[29];
    const float* fc1_b  = (const float*)d_in[30];
    const float* fc2_w  = (const float*)d_in[31];
    const float* fc2_b  = (const float*)d_in[32];
    const float* cls_w  = (const float*)d_in[33];
    const float* cls_b  = (const float*)d_in[34];
    const float* bbox_w = (const float*)d_in[35];
    const float* bbox_b = (const float*)d_in[36];

    static const int CIN[13]  = {3, 64, 64, 128, 128, 256, 256, 256, 512, 512, 512, 512, 512};
    static const int COUT[13] = {64, 64, 128, 128, 256, 256, 256, 512, 512, 512, 512, 512, 512};
    static const int HWs[13]  = {224, 224, 112, 112, 56, 56, 56, 28, 28, 28, 14, 14, 14};
    static const int POOL[13] = {0, 1, 0, 1, 0, 0, 1, 0, 0, 1, 0, 0, 0};

    char* ws = (char*)d_ws;
    size_t o = 0;
    auto alloc = [&](size_t bytes) {
        o = (o + 255) & ~(size_t)255;
        size_t r = o; o += bytes; return r;
    };

    size_t wofs[13];
    int Kpad[13];
    for (int i = 0; i < 13; ++i) {
        Kpad[i] = (i == 0) ? 32 : 9 * CIN[i];
        wofs[i] = alloc((size_t)COUT[i] * Kpad[i] * sizeof(_Float16));
    }
    const size_t actBytes = (size_t)4 * 226 * 226 * 64 * sizeof(_Float16);  // largest padded act
    size_t bufA  = alloc(actBytes);
    size_t bufB  = alloc(actBytes);
    size_t poolO = alloc((size_t)128 * 25088 * sizeof(_Float16));
    size_t h1O   = alloc((size_t)128 * 4096 * sizeof(_Float16));
    size_t h2O   = alloc((size_t)128 * 4096 * sizeof(_Float16));

    // 1) repack conv weights (lane-order, channel-innermost per tap, f16)
    for (int i = 0; i < 13; ++i) {
        size_t tot = (size_t)COUT[i] * Kpad[i];
        pack_wgt_f16<<<(unsigned)((tot + 255) / 256), 256, 0, stream>>>(
            conv_w[i], (_Float16*)(ws + wofs[i]), COUT[i], CIN[i], Kpad[i]);
    }

    auto border = [&](size_t off, int C, int Hp) {
        size_t tot = (size_t)4 * 4 * Hp * C;
        zero_border_nhwc<<<(unsigned)((tot + 255) / 256), 256, 0, stream>>>(
            (_Float16*)(ws + off), C, Hp);
    };

    // 2) input image -> padded NHWC f16
    border(bufA, 3, 226);
    {
        size_t tot = (size_t)12 * 224 * 224;
        cvt_pad_in<<<(unsigned)((tot + 255) / 256), 256, 0, stream>>>(
            inp, (_Float16*)(ws + bufA));
    }

    // 3) VGG16 feature stack (padded NHWC f16, ping-pong buffers)
    size_t cur = bufA, nxt = bufB;
    for (int i = 0; i < 13; ++i) {
        const int H = HWs[i];
        const int total = 4 * H * H;
        const int pixTiles = total / 16;
        border(nxt, COUT[i], H + 2);
        if (i == 0) {
            dim3 grid((unsigned)((pixTiles + 7) / 8), 4, 1);
            conv0_wmma<<<grid, 256, 0, stream>>>(
                (const _Float16*)(ws + cur), (const _Float16*)(ws + wofs[0]), conv_b[0],
                (_Float16*)(ws + nxt));
        } else {
            dim3 grid((unsigned)((pixTiles + 7) / 8), (unsigned)(COUT[i] / 32), 1);
            const _Float16* a  = (const _Float16*)(ws + cur);
            const _Float16* wg = (const _Float16*)(ws + wofs[i]);
            _Float16*       ot = (_Float16*)(ws + nxt);
            const int ci = CIN[i];
            if      (ci == 64  && H == 224) conv3x3_wmma_t<64, 224><<<grid, 256, 0, stream>>>(a, wg, conv_b[i], ot, COUT[i]);
            else if (ci == 64  && H == 112) conv3x3_wmma_t<64, 112><<<grid, 256, 0, stream>>>(a, wg, conv_b[i], ot, COUT[i]);
            else if (ci == 128 && H == 112) conv3x3_wmma_t<128, 112><<<grid, 256, 0, stream>>>(a, wg, conv_b[i], ot, COUT[i]);
            else if (ci == 128 && H == 56)  conv3x3_wmma_t<128, 56><<<grid, 256, 0, stream>>>(a, wg, conv_b[i], ot, COUT[i]);
            else if (ci == 256 && H == 56)  conv3x3_wmma_t<256, 56><<<grid, 256, 0, stream>>>(a, wg, conv_b[i], ot, COUT[i]);
            else if (ci == 256 && H == 28)  conv3x3_wmma_t<256, 28><<<grid, 256, 0, stream>>>(a, wg, conv_b[i], ot, COUT[i]);
            else if (ci == 512 && H == 28)  conv3x3_wmma_t<512, 28><<<grid, 256, 0, stream>>>(a, wg, conv_b[i], ot, COUT[i]);
            else                            conv3x3_wmma_t<512, 14><<<grid, 256, 0, stream>>>(a, wg, conv_b[i], ot, COUT[i]);
        }
        { size_t t = cur; cur = nxt; nxt = t; }
        if (POOL[i]) {
            border(nxt, COUT[i], H / 2 + 2);
            size_t tot = (size_t)4 * (H / 2) * (H / 2) * COUT[i];
            maxpool2_nhwc<<<(unsigned)((tot + 255) / 256), 256, 0, stream>>>(
                (const _Float16*)(ws + cur), (_Float16*)(ws + nxt), COUT[i], H);
            { size_t t = cur; cur = nxt; nxt = t; }
        }
    }

    // 4) ROI adaptive max-pool -> [128, 25088] f16
    roi_pool_k<<<12544, 256, 0, stream>>>(
        (const _Float16*)(ws + cur), rois, ridx, (_Float16*)(ws + poolO));

    // 5) FC head (HBM-bound)
    fc_wmma<<<(4096 / 16) * (128 / 16) / 8, 256, 0, stream>>>(
        fc1_w, (const _Float16*)(ws + poolO), fc1_b, (_Float16*)(ws + h1O), 4096, 25088, 128, 1);
    fc_wmma<<<(4096 / 16) * (128 / 16) / 8, 256, 0, stream>>>(
        fc2_w, (const _Float16*)(ws + h1O), fc2_b, (_Float16*)(ws + h2O), 4096, 4096, 128, 1);

    // 6) heads -> d_out
    heads_k<<<96, 256, 0, stream>>>(
        (const _Float16*)(ws + h2O), cls_w, cls_b, bbox_w, bbox_b, (float*)d_out);
}
